// MultiSelfAttention_37014028157469
// MI455X (gfx1250) — compile-verified, matrix operand order checked by reading, not run on hardware
//
#include <hip/hip_runtime.h>
#include <hip/hip_bf16.h>

typedef __bf16 bf16_t;
typedef __attribute__((ext_vector_type(8)))  __bf16 v8bf;
typedef __attribute__((ext_vector_type(16))) __bf16 v16bf;
typedef __attribute__((ext_vector_type(8)))  float  v8f;
typedef __attribute__((ext_vector_type(4)))  unsigned v4u;
typedef __attribute__((ext_vector_type(8)))  int    v8i;
typedef __attribute__((ext_vector_type(4)))  int    v4i;

// ---------------- helpers ----------------

static __device__ __forceinline__ unsigned short f2bfu(float f) {
    unsigned u = __builtin_bit_cast(unsigned, f);
    u += 0x7FFFu + ((u >> 16) & 1u);        // round-to-nearest-even
    return (unsigned short)(u >> 16);
}
static __device__ __forceinline__ bf16_t f2bf(float f) {
    return __builtin_bit_cast(bf16_t, f2bfu(f));
}

// Load one 16x32 bf16 WMMA A/B fragment from an LDS tile.
// Layout per CDNA5 ISA 7.12.2 (16-bit A-matrix 16x32):
//   lanes 0-15 : row = lane,    K = {0..7, 16..23}  (+kbase)
//   lanes16-31 : row = lane-16, K = {8..15, 24..31} (+kbase)
static __device__ __forceinline__ v16bf load_frag(const bf16_t* base, int row0,
                                                  int ld, int kbase, int lane) {
    const int half = (lane >> 4) & 1;
    const int r    = lane & 15;
    const bf16_t* p = base + (row0 + r) * ld + kbase + half * 8;
    v8bf lo = *(const v8bf*)(p);
    v8bf hi = *(const v8bf*)(p + 16);
    return __builtin_shufflevector(lo, hi, 0,1,2,3,4,5,6,7,8,9,10,11,12,13,14,15);
}

// ---------------- Tensor Data Mover: 2D tile load (global -> LDS) ----------------
// Packs D# group0/group1 per CDNA5 ISA 8.3/8.4 and issues TENSOR_LOAD_TO_LDS.
// data_size = 4 bytes (DWORD granules). tensor_dim == tile_dim (in-bounds by
// construction, OOB logic never fires). LDS row padding done by the TDM itself:
// pad_amount DWORDs inserted after every 2^(pad_interval_code+1) DWORDs.
static __device__ __forceinline__ void tdm_load_2d(const void* gptr, unsigned ldsOff,
                                                   unsigned tileD0Dw, unsigned tileD1,
                                                   unsigned strideDw,
                                                   unsigned padIntervalCode,
                                                   unsigned padAmountCode) {
    unsigned long long ga = (unsigned long long)(size_t)gptr;
    v4u g0;
    g0.x = 1u;                                          // count=1, user mode
    g0.y = ldsOff;                                      // LDS byte address
    g0.z = (unsigned)ga;                                // global addr [31:0]
    g0.w = (unsigned)((ga >> 32) & 0x1FFFFFFu) | (2u << 30);  // addr[56:32] | type=2
    v8i g1;
    g1[0] = (int)((2u << 16) | (1u << 20) |
                  (padIntervalCode << 22) | (padAmountCode << 25)); // dsz=4B, pad_en
    g1[1] = (int)(tileD0Dw << 16);                      // tensor_dim0[15:0]
    g1[2] = (int)(tileD1 << 16);                        // t_dim0 hi=0 | tensor_dim1 lo
    g1[3] = (int)(tileD0Dw << 16);                      // t_dim1 hi=0 | tile_dim0
    g1[4] = (int)(tileD1 & 0xFFFFu);                    // tile_dim1 | tile_dim2=0
    g1[5] = (int)strideDw;                              // tensor_dim0_stride lo32
    g1[6] = 0;                                          // stride0 hi | stride1 lo
    g1[7] = 0;
    v4i z4 = {0, 0, 0, 0};
    v8i z8 = {0, 0, 0, 0, 0, 0, 0, 0};
    __builtin_amdgcn_tensor_load_to_lds(g0, g1, z4, z4, z8, 0);
}

static __device__ __forceinline__ unsigned lds_off(const void* p) {
    return (unsigned)(size_t)p;                         // LDS generic addr: low 32 bits
}

// ---------------- precision-conversion kernels ----------------

__global__ __launch_bounds__(256)
void cast_f32_to_bf16_kernel(const float* __restrict__ in, bf16_t* __restrict__ out, int n) {
    int idx = (blockIdx.x * 256 + threadIdx.x) * 4;
    if (idx + 3 < n) {
        float4 v = *(const float4*)(in + idx);
        out[idx + 0] = f2bf(v.x);
        out[idx + 1] = f2bf(v.y);
        out[idx + 2] = f2bf(v.z);
        out[idx + 3] = f2bf(v.w);
    }
}

// W (KxN, f32, row-major) -> Wt (NxK, bf16, row-major)
__global__ __launch_bounds__(256)
void transpose_f32_to_bf16_kernel(const float* __restrict__ W, bf16_t* __restrict__ Wt,
                                  int K, int N) {
    __shared__ float tile[32][33];
    const int tx = threadIdx.x & 31;
    const int ty = threadIdx.x >> 5;
    const int n0 = blockIdx.x * 32;
    const int k0 = blockIdx.y * 32;
#pragma unroll
    for (int i = 0; i < 4; ++i)
        tile[ty + i * 8][tx] = W[(size_t)(k0 + ty + i * 8) * N + n0 + tx];
    __syncthreads();
#pragma unroll
    for (int i = 0; i < 4; ++i)
        Wt[(size_t)(n0 + ty + i * 8) * K + k0 + tx] = f2bf(tile[tx][ty + i * 8]);
}

// ---------------- WMMA GEMM:  C[M,N] = A[M,K] * Bt[N,K]^T + bias ----------------
// TDM-staged, double-buffered LDS. 128x128 block tile, 256 threads (8 waves),
// each wave computes 32x64 (2x4 WMMA tiles). K step 32. Wave 0 drives the DMA:
// issue tiles k+1, wait TENSORcnt for tiles k, barrier, everyone WMMAs.

template <bool OUT_BF16>
__global__ __launch_bounds__(256)
void gemm_bf16_kernel(const bf16_t* __restrict__ A, const bf16_t* __restrict__ Bt,
                      const float* __restrict__ bias, void* __restrict__ Cout,
                      int M, int N, int K) {
    constexpr int LD = 40;                    // 32 + 8 pad (TDM pads 4 DW per 16 DW)
    __shared__ bf16_t Alds[2][128 * LD];
    __shared__ bf16_t Blds[2][128 * LD];

    const int tid  = threadIdx.x;
    const int lane = tid & 31;
    const int w    = tid >> 5;
    const int wr   = w & 3;
    const int wc   = w >> 2;
    const int m0   = blockIdx.y * 128;
    const int n0   = blockIdx.x * 128;
    const int nK   = K >> 5;

    v8f acc[2][4] = {};

    const unsigned strideDw = (unsigned)(K >> 1);
    if (tid < 32) {   // wave 0 primes buffer 0
        tdm_load_2d(A  + (size_t)m0 * K, lds_off(&Alds[0][0]), 16u, 128u, strideDw, 3u, 3u);
        tdm_load_2d(Bt + (size_t)n0 * K, lds_off(&Blds[0][0]), 16u, 128u, strideDw, 3u, 3u);
    }

    for (int ki = 0; ki < nK; ++ki) {
        const int cur = ki & 1;
        if (tid < 32) {
            if (ki + 1 < nK) {
                const int nxt = (ki + 1) & 1;
                const size_t ko = (size_t)(ki + 1) << 5;
                tdm_load_2d(A  + (size_t)m0 * K + ko, lds_off(&Alds[nxt][0]),
                            16u, 128u, strideDw, 3u, 3u);
                tdm_load_2d(Bt + (size_t)n0 * K + ko, lds_off(&Blds[nxt][0]),
                            16u, 128u, strideDw, 3u, 3u);
                __builtin_amdgcn_s_wait_tensorcnt(2);   // current pair complete
            } else {
                __builtin_amdgcn_s_wait_tensorcnt(0);
            }
        }
        __syncthreads();

        const bf16_t* Ab = Alds[cur];
        const bf16_t* Bb = Blds[cur];
        v16bf aF[2], bF[4];
#pragma unroll
        for (int i = 0; i < 2; ++i) aF[i] = load_frag(Ab, wr * 32 + 16 * i, LD, 0, lane);
#pragma unroll
        for (int j = 0; j < 4; ++j) bF[j] = load_frag(Bb, wc * 64 + 16 * j, LD, 0, lane);
#pragma unroll
        for (int i = 0; i < 2; ++i)
#pragma unroll
            for (int j = 0; j < 4; ++j)
                acc[i][j] = __builtin_amdgcn_wmma_f32_16x16x32_bf16(
                    false, aF[i], false, bF[j], (short)0, acc[i][j], false, false);
        __syncthreads();
    }

    const int half = (lane >> 4) & 1;
    const int ln   = lane & 15;
#pragma unroll
    for (int i = 0; i < 2; ++i) {
#pragma unroll
        for (int j = 0; j < 4; ++j) {
            const int col = n0 + wc * 64 + 16 * j + ln;
            const float bia = bias[col];
#pragma unroll
            for (int r = 0; r < 8; ++r) {
                const int row = m0 + wr * 32 + 16 * i + r + 8 * half;
                const float v = acc[i][j][r] + bia;
                if (OUT_BF16) ((bf16_t*)Cout)[(size_t)row * N + col] = f2bf(v);
                else          ((float*) Cout)[(size_t)row * N + col] = v;
            }
        }
    }
}

// ---------------- Flash attention ----------------
// qkv: (B*T, 6144) bf16; q at col h*128, k at 2048+h*128, v at 4096+h*128.
// One workgroup = (b,h) x 64-row q tile; 4 waves; wave handles 16 q rows.
// K tiles stream via TDM, double-buffered; V is transposed manually into LDS.

__global__ __launch_bounds__(128)
void flash_attn_kernel(const bf16_t* __restrict__ qkv, bf16_t* __restrict__ y) {
    constexpr int LDQ = 136;                  // 128 + 8 (TDM pads 4 DW per 64 DW)
    constexpr int LDV = 72;                   // 64 + 8 pad
    __shared__ bf16_t Qs[64 * LDQ];
    __shared__ bf16_t Ks[2][64 * LDQ];
    __shared__ bf16_t Vs[128 * LDV];          // transposed: Vs[d][key]
    __shared__ bf16_t Ps[64 * LDV];

    const int tid  = threadIdx.x;
    const int lane = tid & 31;
    const int wv   = tid >> 5;
    const int bh   = blockIdx.x;
    const int qt   = blockIdx.y;
    const int b    = bh >> 4;
    const int h    = bh & 15;
    const int q0   = qt * 64;
    const size_t rowBase = (size_t)b * 1024;

    // ---- load Q tile; prime K DMA for tile 0 ----
#pragma unroll
    for (int i = 0; i < 8; ++i) {
        int c = tid + 128 * i;
        int row = c >> 4, kc = (c & 15) * 8;
        *(v8bf*)&Qs[row * LDQ + kc] =
            *(const v8bf*)&qkv[(rowBase + q0 + row) * 6144 + h * 128 + kc];
    }
    if (tid < 32)
        tdm_load_2d(qkv + rowBase * 6144 + 2048 + h * 128,
                    lds_off(&Ks[0][0]), 64u, 64u, 3072u, 5u, 3u);
    __syncthreads();
    v16bf qf[4];
#pragma unroll
    for (int kc = 0; kc < 4; ++kc) qf[kc] = load_frag(Qs, wv * 16, LDQ, kc * 32, lane);

    v8f o[8] = {};
    float m_[8], l_[8];
#pragma unroll
    for (int r = 0; r < 8; ++r) { m_[r] = -1e30f; l_[r] = 0.0f; }

    const int   half = (lane >> 4) & 1;
    const int   ln   = lane & 15;
    const float sc   = 0.08838834764831845f;  // 1/sqrt(128)

    for (int j = 0; j <= qt; ++j) {
        // ---- DMA K tile j+1; stage V tile j (transposed) with vector loads ----
        if (tid < 32 && j < qt)
            tdm_load_2d(qkv + (rowBase + (size_t)(j + 1) * 64) * 6144 + 2048 + h * 128,
                        lds_off(&Ks[(j + 1) & 1][0]), 64u, 64u, 3072u, 5u, 3u);
#pragma unroll
        for (int i = 0; i < 8; ++i) {
            int c = tid + 128 * i;
            int row = c >> 4, dc = (c & 15) * 8;
            v8bf vv = *(const v8bf*)&qkv[(rowBase + j * 64 + row) * 6144 + 4096 + h * 128 + dc];
#pragma unroll
            for (int e = 0; e < 8; ++e) Vs[(dc + e) * LDV + row] = vv[e];
        }
        if (tid < 32) {
            if (j < qt) __builtin_amdgcn_s_wait_tensorcnt(1);
            else        __builtin_amdgcn_s_wait_tensorcnt(0);
        }
        __syncthreads();

        // ---- S = Q K^T (wave: 16 x 64, fp32) ----
        const bf16_t* Kb = Ks[j & 1];
        v8f s[4] = {};
#pragma unroll
        for (int kc = 0; kc < 4; ++kc) {
#pragma unroll
            for (int nt = 0; nt < 4; ++nt) {
                v16bf kf = load_frag(Kb, nt * 16, LDQ, kc * 32, lane);
                s[nt] = __builtin_amdgcn_wmma_f32_16x16x32_bf16(
                    false, qf[kc], false, kf, (short)0, s[nt], false, false);
            }
        }

        // ---- scale, causal mask, online softmax, stash P (bf16) in LDS ----
        const bool diag = (j == qt);
#pragma unroll
        for (int r = 0; r < 8; ++r) {
            const int qrow = q0 + wv * 16 + r + 8 * half;
            float v[4];
#pragma unroll
            for (int nt = 0; nt < 4; ++nt) {
                const int kcol = j * 64 + nt * 16 + ln;
                float x = s[nt][r] * sc;
                if (diag && kcol > qrow) x = -1e30f;
                v[nt] = x;
            }
            float mloc = fmaxf(fmaxf(v[0], v[1]), fmaxf(v[2], v[3]));
#pragma unroll
            for (int off = 1; off < 16; off <<= 1)
                mloc = fmaxf(mloc, __shfl_xor(mloc, off, 32));
            const float mnew = fmaxf(m_[r], mloc);
            const float corr = __expf(m_[r] - mnew);
            float psum = 0.0f;
#pragma unroll
            for (int nt = 0; nt < 4; ++nt) {
                const float p = __expf(v[nt] - mnew);
                psum += p;
                Ps[(wv * 16 + r + 8 * half) * LDV + nt * 16 + ln] = f2bf(p);
            }
#pragma unroll
            for (int off = 1; off < 16; off <<= 1)
                psum += __shfl_xor(psum, off, 32);
            l_[r] = l_[r] * corr + psum;
            m_[r] = mnew;
#pragma unroll
            for (int dt = 0; dt < 8; ++dt) o[dt][r] *= corr;
        }
        __syncthreads();

        // ---- O += P V  (A = P 16x64, B = Vs[d][key]) ----
#pragma unroll
        for (int kk = 0; kk < 2; ++kk) {
            v16bf pf = load_frag(Ps, wv * 16, LDV, kk * 32, lane);
#pragma unroll
            for (int dt = 0; dt < 8; ++dt) {
                v16bf vf = load_frag(Vs, dt * 16, LDV, kk * 32, lane);
                o[dt] = __builtin_amdgcn_wmma_f32_16x16x32_bf16(
                    false, pf, false, vf, (short)0, o[dt], false, false);
            }
        }
        __syncthreads();
    }

    // ---- normalize and write y (bf16, (B*T, 2048) layout) ----
#pragma unroll
    for (int r = 0; r < 8; ++r) {
        const float inv  = 1.0f / l_[r];
        const int   trow = q0 + wv * 16 + r + 8 * half;
#pragma unroll
        for (int dt = 0; dt < 8; ++dt)
            y[(rowBase + trow) * 2048 + h * 128 + dt * 16 + ln] = f2bf(o[dt][r] * inv);
    }
}

// ---------------- launch ----------------

extern "C" void kernel_launch(void* const* d_in, const int* in_sizes, int n_in,
                              void* d_out, int out_size, void* d_ws, size_t ws_size,
                              hipStream_t stream) {
    (void)in_sizes; (void)n_in; (void)out_size; (void)ws_size;

    const float* x      = (const float*)d_in[0];   // (4,1024,2048)
    const float* w_attn = (const float*)d_in[1];   // (2048,6144)
    const float* b_attn = (const float*)d_in[2];   // (6144)
    const float* w_proj = (const float*)d_in[3];   // (2048,2048)
    const float* b_proj = (const float*)d_in[4];   // (2048)
    float*       out    = (float*)d_out;           // (4,1024,2048)

    char* ws = (char*)d_ws;
    bf16_t* xb   = (bf16_t*)(ws);                          // 16 MB  (4096x2048)
    bf16_t* wat  = (bf16_t*)(ws + 16777216);               // 24 MB  (6144x2048) transposed
    bf16_t* wpt  = (bf16_t*)(ws + 41943040);               //  8 MB  (2048x2048) transposed
    bf16_t* qkvb = (bf16_t*)(ws + 50331648);               // 48 MB  (4096x6144)
    bf16_t* yb   = (bf16_t*)(ws + 100663296);              // 16 MB  (4096x2048)

    cast_f32_to_bf16_kernel<<<8192, 256, 0, stream>>>(x, xb, 4096 * 2048);
    transpose_f32_to_bf16_kernel<<<dim3(192, 64), 256, 0, stream>>>(w_attn, wat, 2048, 6144);
    transpose_f32_to_bf16_kernel<<<dim3(64, 64),  256, 0, stream>>>(w_proj, wpt, 2048, 2048);

    // qkv = x @ w_attn + b_attn  (store bf16)
    gemm_bf16_kernel<true><<<dim3(48, 32), 256, 0, stream>>>(
        xb, wat, b_attn, (void*)qkvb, 4096, 6144, 2048);

    // causal flash attention -> y (bf16)
    flash_attn_kernel<<<dim3(64, 16), 128, 0, stream>>>(qkvb, yb);

    // out = y @ w_proj + b_proj  (store fp32)
    gemm_bf16_kernel<false><<<dim3(16, 32), 256, 0, stream>>>(
        yb, wpt, b_proj, (void*)out, 4096, 2048, 2048);
}
